// Attention_86586540687646
// MI455X (gfx1250) — matrix operand order repaired and checked by reading, not compiled, not run on hardware
//
#include <hip/hip_runtime.h>
#include <stdint.h>

// Problem constants
#define HW   2809   // 53*53
#define WW   53
#define CC   512
#define NH   8
#define DH   64

typedef __attribute__((ext_vector_type(16))) __bf16 v16bf;
typedef __attribute__((ext_vector_type(8)))  float  v8f;
typedef int v4i __attribute__((ext_vector_type(4)));
typedef float f4a __attribute__((ext_vector_type(4), aligned(4)));

union Frag { v16bf v; uint32_t u[8]; };

// ---- f32 -> bf16 (RNE) helpers ------------------------------------------
#if __has_builtin(__builtin_amdgcn_cvt_pk_bf16_f32)
__device__ __forceinline__ uint32_t pk2bf(float a, float b) {
  typedef __attribute__((ext_vector_type(2))) __bf16 v2bf;
  union { v2bf v; uint32_t u; } t;
  t.v = __builtin_amdgcn_cvt_pk_bf16_f32(a, b);
  return t.u;
}
#else
// round both, then merge hi-halves with one v_perm_b32 (no 16-bit movs)
__device__ __forceinline__ uint32_t pk2bf(float a, float b) {
  uint32_t ua = __float_as_uint(a), ub = __float_as_uint(b);
  ua += 0x7FFFu + ((ua >> 16) & 1u);
  ub += 0x7FFFu + ((ub >> 16) & 1u);
  return __builtin_amdgcn_perm(ub, ua, 0x07060302u);  // {ub.b3,ub.b2,ua.b3,ua.b2}
}
#endif

__device__ __forceinline__ uint32_t f2bf1(float a) {  // low 16 bits valid
  uint32_t ua = __float_as_uint(a);
  ua += 0x7FFFu + ((ua >> 16) & 1u);
  return ua >> 16;
}

// ---- async global->LDS copy (ASYNCcnt path) ------------------------------
#if __has_builtin(__builtin_amdgcn_global_load_async_to_lds_b128)
#define ASYNC_B128(g, l)                                                    \
  __builtin_amdgcn_global_load_async_to_lds_b128(                           \
      (__attribute__((address_space(1))) v4i*)(g),                          \
      (__attribute__((address_space(3))) v4i*)(l), 0, 0)
#else
#define ASYNC_B128(g, l)                                                    \
  asm volatile("global_load_async_to_lds_b128 %0, %1, off"                  \
               :: "v"((uint32_t)(uintptr_t)(l)), "v"((const void*)(g))      \
               : "memory")
#endif

#if __has_builtin(__builtin_amdgcn_s_wait_asynccnt)
#define WAIT_ASYNC0() __builtin_amdgcn_s_wait_asynccnt(0)
#else
#define WAIT_ASYNC0() asm volatile("s_wait_asynccnt 0x0" ::: "memory")
#endif

// ---------------------------------------------------------------------------
// Kernel 1: qkv = w_in(1536x512) @ x(512xHW) per batch -> bf16 ws in
// [b][head][hw][d] layout, with the positional fold k' = 8k + emb/8 so the
// attention kernel only needs sim = q . k'.
// Block: 128 threads = 4 waves; block tile 64(o) x 16(hw); wave tile 16x16.
// ---------------------------------------------------------------------------
__global__ __launch_bounds__(128)
void qkv_proj_kernel(const float* __restrict__ x, const float* __restrict__ w_in,
                     const float* __restrict__ pos_h, const float* __restrict__ pos_w,
                     unsigned short* __restrict__ qw, unsigned short* __restrict__ kw,
                     unsigned short* __restrict__ vw) {
  __shared__ unsigned short smA[64][32];  // w_in tile [o_local][k_local]
  __shared__ unsigned short smB[16][32];  // x tile transposed: [hw_local][k_local]

  const int tid = threadIdx.x;
  const int L = tid & 31, wv = tid >> 5;
  const int m = L & 15, g = L >> 4;
  const int o0  = blockIdx.x * 64;
  const int hw0 = blockIdx.y * 16;
  const int b   = blockIdx.z;

  v8f acc;
  for (int r = 0; r < 8; ++r) acc[r] = 0.f;

  for (int kc = 0; kc < CC; kc += 32) {
    __syncthreads();
    // A tile: float4 loads, packed-bf16 b32 stores
    for (int it = 0; it < 4; ++it) {
      int idx = tid + it * 128;                 // 512 float4 slots
      int row = idx >> 3, c4 = (idx & 7) * 4;
      float4 f = *(const float4*)&w_in[(size_t)(o0 + row) * CC + kc + c4];
      *(uint32_t*)&smA[row][c4]     = pk2bf(f.x, f.y);
      *(uint32_t*)&smA[row][c4 + 2] = pk2bf(f.z, f.w);
    }
    // B tile: two k-rows per thread, packed to one b32 in transposed layout
    for (int it = 0; it < 2; ++it) {
      int idx = tid + it * 128;                 // 256 (n, k-pair) slots
      int n = idx & 15, kp = idx >> 4;
      int hw = hw0 + n;
      float f0 = 0.f, f1 = 0.f;
      if (hw < HW) {
        const float* px = &x[((size_t)b * CC + kc + 2 * kp) * HW + hw];
        f0 = px[0];
        f1 = px[HW];
      }
      *(uint32_t*)&smB[n][2 * kp] = pk2bf(f0, f1);
    }
    __syncthreads();

    // Gather WMMA fragments per documented CDNA5 layouts
    Frag a, bfr;
    for (int v = 0; v < 8; ++v) {
      int ka = (v >> 2) * 16 + g * 8 + (v & 3) * 2;      // A 16x32 bf16 layout
      a.u[v] = *(const uint32_t*)&smA[wv * 16 + m][ka];
      int kb = (v >> 2) * 16 + g * 8 + (v & 3) * 2;      // B 32x16 bf16 layout
      bfr.u[v] = *(const uint32_t*)&smB[m][kb];
    }
    acc = __builtin_amdgcn_wmma_f32_16x16x32_bf16(false, a.v, false, bfr.v,
                                                  (short)0, acc, false, false);
  }

  // Epilogue: per lane, 8 consecutive d elements -> one packed b128 store.
  const int sec  = o0 / CC;               // 0=q 1=k 2=v (64-aligned, never spans)
  const int head = (o0 % CC) / DH;
  const int hw   = hw0 + m;               // C layout: col N = lane%16
  if (hw < HW) {
    const int dd0 = wv * 16 + 8 * g;      // rows M = r + 8g -> dd = dd0 + r
    float vals[8];
    for (int r = 0; r < 8; ++r) vals[r] = acc[r];
    if (sec == 1) {
      const int ih = hw / WW, iw = hw % WW;
      float4 ph0 = *(const float4*)&pos_h[ih * DH + dd0];
      float4 ph1 = *(const float4*)&pos_h[ih * DH + dd0 + 4];
      float4 pw0 = *(const float4*)&pos_w[iw * DH + dd0];
      float4 pw1 = *(const float4*)&pos_w[iw * DH + dd0 + 4];
      const float e[8] = {ph0.x + pw0.x, ph0.y + pw0.y, ph0.z + pw0.z, ph0.w + pw0.w,
                          ph1.x + pw1.x, ph1.y + pw1.y, ph1.z + pw1.z, ph1.w + pw1.w};
      for (int r = 0; r < 8; ++r) vals[r] = 8.0f * vals[r] + 0.125f * e[r];
    }
    uint4 pk;
    pk.x = pk2bf(vals[0], vals[1]);
    pk.y = pk2bf(vals[2], vals[3]);
    pk.z = pk2bf(vals[4], vals[5]);
    pk.w = pk2bf(vals[6], vals[7]);
    unsigned short* dst = (sec == 0) ? qw : (sec == 1 ? kw : vw);
    size_t idx0 = (((size_t)(b * NH + head)) * HW + hw) * DH + dd0;  // 16B aligned
    *(uint4*)&dst[idx0] = pk;
  }
}

// ---------------------------------------------------------------------------
// Kernel 2: flash attention. 256 threads = 8 waves, Br=128 query rows,
// K'/V streamed in 64-row chunks. Q/K staged with async global->LDS b128;
// V transposed through registers with v_perm_b32.
// ---------------------------------------------------------------------------
#define NCHUNK ((HW + 63) / 64)   // 44

__global__ __launch_bounds__(256)
void attn_kernel(const unsigned short* __restrict__ qw,
                 const unsigned short* __restrict__ kw,
                 const unsigned short* __restrict__ vw,
                 float* __restrict__ out) {
  __shared__ unsigned short smQ[128][64];     // [i_local][d]
  __shared__ unsigned short smK[64][64];      // [j_local][d]
  __shared__ unsigned short smV[64][64];      // transposed: [d][j_local]
  __shared__ unsigned short smP[8][16][64];   // per-wave P tile [i_local][j_local]

  const int tid = threadIdx.x;
  const int L = tid & 31, wv = tid >> 5;
  const int m = L & 15, g = L >> 4;
  const int i0   = blockIdx.x * 128;
  const int head = blockIdx.y;
  const int b    = blockIdx.z;
  const size_t base = ((size_t)(b * NH + head)) * HW * DH;

  // Stage Q block: async b128 copies (clamped rows for the ragged tail)
  for (int it = 0; it < 4; ++it) {
    int idx = tid + it * 256;                // 1024 x 16B = 16KB
    int row = idx >> 3, c8 = (idx & 7) * 8;
    int rg = i0 + row; if (rg > HW - 1) rg = HW - 1;
    ASYNC_B128(qw + base + (size_t)rg * DH + c8, &smQ[row][c8]);
  }
  WAIT_ASYNC0();
  __syncthreads();

  // Q A-fragments (K = d = 64 -> two 16x32 fragments), kept in VGPRs
  Frag aq[2];
  for (int f = 0; f < 2; ++f)
    for (int v = 0; v < 8; ++v) {
      int kk = 32 * f + (v >> 2) * 16 + g * 8 + (v & 3) * 2;
      aq[f].u[v] = *(const uint32_t*)&smQ[wv * 16 + m][kk];
    }

  v8f acc_o[4];
  for (int s4 = 0; s4 < 4; ++s4)
    for (int r = 0; r < 8; ++r) acc_o[s4][r] = 0.f;
  float mrow[8], lrow[8];
  for (int r = 0; r < 8; ++r) { mrow[r] = -3.0e38f; lrow[r] = 0.f; }

  const uint32_t* vs = (const uint32_t*)(vw + base);

  for (int jc = 0; jc < NCHUNK; ++jc) {
    const int j0 = jc * 64;
    __syncthreads();
    // K chunk: async b128 global->LDS (natural [j][d] layout)
    for (int it = 0; it < 2; ++it) {
      int idx = tid + it * 256;              // 512 x 16B = 8KB
      int jl = idx >> 3, c8 = (idx & 7) * 8;
      int jg = j0 + jl; if (jg > HW - 1) jg = HW - 1;
      ASYNC_B128(kw + base + (size_t)jg * DH + c8, &smK[jl][c8]);
    }
    // V chunk transposed: 2x2 bf16 block per task via v_perm_b32
    for (int it = 0; it < 4; ++it) {
      int idx = tid + it * 256;              // 1024 (j-pair, d-pair) tasks
      int jp = idx >> 5, cp = idx & 31;
      int jl = 2 * jp;
      int jg0 = j0 + jl;     if (jg0 > HW - 1) jg0 = HW - 1;
      int jg1 = j0 + jl + 1; if (jg1 > HW - 1) jg1 = HW - 1;
      uint32_t w0 = vs[(size_t)jg0 * 32 + cp];
      uint32_t w1 = vs[(size_t)jg1 * 32 + cp];
      *(uint32_t*)&smV[2 * cp][jl]     = __builtin_amdgcn_perm(w1, w0, 0x05040100u);
      *(uint32_t*)&smV[2 * cp + 1][jl] = __builtin_amdgcn_perm(w1, w0, 0x07060302u);
    }
    WAIT_ASYNC0();
    __syncthreads();

    // Prefetch next chunk toward the WGP while we compute on this one
    if (jc + 1 < NCHUNK) {
      int pj = j0 + 64 + (tid & 63);
      if (pj < HW) {
        __builtin_prefetch(kw + base + (size_t)pj * DH, 0, 0);
        __builtin_prefetch(vw + base + (size_t)pj * DH, 0, 0);
      }
    }

    // S = Q . K'^T  (4 j-subtiles x 2 K-halves)
    v8f s[4];
    for (int t = 0; t < 4; ++t) {
      Frag bk0, bk1;
      for (int v = 0; v < 8; ++v) {
        int k0 = (v >> 2) * 16 + g * 8 + (v & 3) * 2;
        bk0.u[v] = *(const uint32_t*)&smK[16 * t + m][k0];
        bk1.u[v] = *(const uint32_t*)&smK[16 * t + m][32 + k0];
      }
      v8f z; for (int r = 0; r < 8; ++r) z[r] = 0.f;
      s[t] = __builtin_amdgcn_wmma_f32_16x16x32_bf16(false, aq[0].v, false, bk0.v,
                                                     (short)0, z, false, false);
      s[t] = __builtin_amdgcn_wmma_f32_16x16x32_bf16(false, aq[1].v, false, bk1.v,
                                                     (short)0, s[t], false, false);
      int j = j0 + 16 * t + m;               // C layout: col N = lane%16
      if (j >= HW)
        for (int r = 0; r < 8; ++r) s[t][r] = -3.0e38f;
    }

    // Online softmax: row reductions within each 16-lane half
    float mnew[8], alpha[8];
    for (int r = 0; r < 8; ++r) {
      float mx = fmaxf(fmaxf(s[0][r], s[1][r]), fmaxf(s[2][r], s[3][r]));
      for (int off = 8; off >= 1; off >>= 1)
        mx = fmaxf(mx, __shfl_xor(mx, off, 32));
      mnew[r]  = fmaxf(mrow[r], mx);
      alpha[r] = __expf(mrow[r] - mnew[r]);
      mrow[r]  = mnew[r];
    }
    // P = exp(S - m), spilled to per-wave LDS tile for operand reshape
    for (int t = 0; t < 4; ++t)
      for (int r = 0; r < 8; ++r) {
        float p = __expf(s[t][r] - mnew[r]);
        s[t][r] = p;
        smP[wv][r + 8 * g][16 * t + m] = (unsigned short)f2bf1(p);
      }
    for (int r = 0; r < 8; ++r) {
      float rs = (s[0][r] + s[1][r]) + (s[2][r] + s[3][r]);
      for (int off = 8; off >= 1; off >>= 1)
        rs += __shfl_xor(rs, off, 32);
      lrow[r] = lrow[r] * alpha[r] + rs;
    }
    for (int s4 = 0; s4 < 4; ++s4)
      for (int r = 0; r < 8; ++r) acc_o[s4][r] *= alpha[r];

    // Reload P as A-fragments (intra-wave DS ordering guarantees visibility)
    Frag ap[2];
    for (int f = 0; f < 2; ++f)
      for (int v = 0; v < 8; ++v) {
        int kk = 32 * f + (v >> 2) * 16 + g * 8 + (v & 3) * 2;
        ap[f].u[v] = *(const uint32_t*)&smP[wv][m][kk];
      }
    // acc_o += P . V   (4 d-subtiles x 2 K-halves)
    for (int s4 = 0; s4 < 4; ++s4) {
      Frag bv0, bv1;
      for (int v = 0; v < 8; ++v) {
        int k0 = (v >> 2) * 16 + g * 8 + (v & 3) * 2;
        bv0.u[v] = *(const uint32_t*)&smV[16 * s4 + m][k0];
        bv1.u[v] = *(const uint32_t*)&smV[16 * s4 + m][32 + k0];
      }
      acc_o[s4] = __builtin_amdgcn_wmma_f32_16x16x32_bf16(false, ap[0].v, false, bv0.v,
                                                          (short)0, acc_o[s4], false, false);
      acc_o[s4] = __builtin_amdgcn_wmma_f32_16x16x32_bf16(false, ap[1].v, false, bv1.v,
                                                          (short)0, acc_o[s4], false, false);
    }
  }

  // Epilogue: out[b][head*64+dd][hw] = acc/l ; 8 consecutive hw per lane
  const int rowbase = i0 + wv * 16 + 8 * g;
  float rl[8];
  for (int r = 0; r < 8; ++r) rl[r] = 1.0f / lrow[r];
  for (int s4 = 0; s4 < 4; ++s4) {
    int dd = 16 * s4 + m;
    float* po = &out[((size_t)b * CC + head * DH + dd) * HW];
    if (rowbase + 7 < HW) {
      f4a v0, v1;
      for (int r = 0; r < 4; ++r) v0[r] = acc_o[s4][r] * rl[r];
      for (int r = 0; r < 4; ++r) v1[r] = acc_o[s4][4 + r] * rl[4 + r];
      *(f4a*)&po[rowbase]     = v0;
      *(f4a*)&po[rowbase + 4] = v1;
    } else {
      for (int r = 0; r < 8; ++r)
        if (rowbase + r < HW) po[rowbase + r] = acc_o[s4][r] * rl[r];
    }
  }
}

// ---------------------------------------------------------------------------
extern "C" void kernel_launch(void* const* d_in, const int* in_sizes, int n_in,
                              void* d_out, int out_size, void* d_ws, size_t ws_size,
                              hipStream_t stream) {
  (void)in_sizes; (void)n_in; (void)out_size; (void)ws_size;
  const float* x     = (const float*)d_in[0];
  const float* w_in  = (const float*)d_in[1];
  const float* pos_h = (const float*)d_in[2];
  const float* pos_w = (const float*)d_in[3];
  float* out = (float*)d_out;

  // bf16 q / k' / v in workspace: 3 * 2*8*2809*64 * 2B ~= 17.3 MB
  const size_t per = (size_t)2 * NH * HW * DH;
  unsigned short* qw = (unsigned short*)d_ws;
  unsigned short* kw = qw + per;
  unsigned short* vw = kw + per;

  dim3 g1(3 * CC / 64, (HW + 15) / 16, 2);
  qkv_proj_kernel<<<g1, 128, 0, stream>>>(x, w_in, pos_h, pos_w, qw, kw, vw);

  dim3 g2((HW + 127) / 128, NH, 2);
  attn_kernel<<<g2, 256, 0, stream>>>(qw, kw, vw, out);
}
